// EncoderLayer_64175401337444
// MI455X (gfx1250) — matrix operand order repaired and checked by reading, not compiled, not run on hardware
//
#include <hip/hip_runtime.h>

#define NTOK 131072
#define CH   256
#define NH   8
#define HD   32
#define KW   128
#define HID  1024
#define SCALE 0.17677669529663687f
#define LN_EPS 1e-5f

typedef __attribute__((ext_vector_type(16))) __bf16 v16bf;
typedef __attribute__((ext_vector_type(8)))  float  v8f;
typedef __attribute__((ext_vector_type(4)))  int    v4i;

#if defined(__gfx1250__) && __has_builtin(__builtin_amdgcn_global_load_async_to_lds_b128) && __has_builtin(__builtin_amdgcn_s_wait_asynccnt)
#define USE_ASYNC_LDS 1
#else
#define USE_ASYNC_LDS 0
#endif

union Frag16 { v16bf v; unsigned short u[16]; };

__device__ __forceinline__ unsigned short f2bf(float f) {
  unsigned int u = __builtin_bit_cast(unsigned int, f);
  u = (u + 0x7FFFu + ((u >> 16) & 1u)) >> 16;
  return (unsigned short)u;
}

// ---- WMMA fragment loaders (bf16 stored as ushort) ----
// A fragment 16x32 (MxK): lane l: m = l&15; element j -> k = (l>>4)*16 + j
__device__ __forceinline__ v16bf load_frag_a(const unsigned short* __restrict__ src,
                                             int ld, int lane) {
  Frag16 f;
  const int m = lane & 15, hi = lane >> 4;
#pragma unroll
  for (int j = 0; j < 16; ++j) {
    const int k = (hi << 4) + j;
    f.u[j] = src[m * ld + k];
  }
  return f.v;
}

// B fragment 32x16 (KxN) where element (k,n) = src[n*ld + k]  (W stored (N,K) row-major)
__device__ __forceinline__ v16bf load_frag_b(const unsigned short* __restrict__ src,
                                             int ld, int lane) {
  Frag16 f;
  const int n = lane & 15, hi = lane >> 4;
#pragma unroll
  for (int j = 0; j < 16; ++j) {
    const int k = (hi << 4) + j;
    f.u[j] = src[n * ld + k];
  }
  return f.v;
}

// B fragment 32x16 (KxN) where element (k,n) = src[k*ld + n]  (K-major storage, V)
__device__ __forceinline__ v16bf load_frag_bT(const unsigned short* __restrict__ src,
                                              int ld, int lane) {
  Frag16 f;
  const int n = lane & 15, hi = lane >> 4;
#pragma unroll
  for (int j = 0; j < 16; ++j) {
    const int k = (hi << 4) + j;
    f.u[j] = src[k * ld + n];
  }
  return f.v;
}

__device__ __forceinline__ v8f wmma_bf16(v16bf a, v16bf b, v8f c) {
  return __builtin_amdgcn_wmma_f32_16x16x32_bf16(false, a, false, b, (short)0, c,
                                                 false, false);
}

// ---------------------------------------------------------------------------
__global__ void __launch_bounds__(256) cvt_bf16_kernel(const float* __restrict__ in,
                                                       unsigned short* __restrict__ out,
                                                       int n) {
  int i = blockIdx.x * 256 + threadIdx.x;
  if (i < n) out[i] = f2bf(in[i]);
}

// ---------------------------------------------------------------------------
// LayerNorm over C=256, one row per 256-thread block, bf16 output
__global__ void __launch_bounds__(256) ln_bf16_kernel(const float* __restrict__ x,
                                                      const float* __restrict__ g,
                                                      const float* __restrict__ b,
                                                      unsigned short* __restrict__ out) {
  __shared__ float red[256];
  const long row = blockIdx.x;
  const int t = threadIdx.x;
  float v = x[row * CH + t];
  red[t] = v;
  __syncthreads();
#pragma unroll
  for (int s = 128; s > 0; s >>= 1) {
    if (t < s) red[t] += red[t + s];
    __syncthreads();
  }
  const float mu = red[0] * (1.0f / CH);
  __syncthreads();
  const float d = v - mu;
  red[t] = d * d;
  __syncthreads();
#pragma unroll
  for (int s = 128; s > 0; s >>= 1) {
    if (t < s) red[t] += red[t + s];
    __syncthreads();
  }
  const float var = red[0] * (1.0f / CH);
  const float y = d * rsqrtf(var + LN_EPS) * g[t] + b[t];
  out[row * CH + t] = f2bf(y);
}

// ---------------------------------------------------------------------------
// GEMM  O = act(A @ B^T + bias [+ resid]);  A: MxK bf16, B: NxK bf16 (row-major)
// MODE 0: store bf16 (qkv) | MODE 1: +resid, store f32 (proj/mlp2) | MODE 2: GeLU, bf16 (mlp1)
// Block: 256 threads = 8 waves; block tile 128(M) x 64(N); wave tile 32x32.
template <int MODE>
__global__ void __launch_bounds__(256) gemm_bf16_kernel(
    const unsigned short* __restrict__ A, const unsigned short* __restrict__ B,
    const float* __restrict__ bias, const float* __restrict__ resid,
    unsigned short* __restrict__ Obf, float* __restrict__ Of, int Nn, int Kd) {
  const int lane = threadIdx.x & 31;
  const int wave = threadIdx.x >> 5;
  const int wm = wave & 3;
  const int wn = wave >> 2;
  const long row0 = (long)blockIdx.y * 128 + wm * 32;
  const int col0 = blockIdx.x * 64 + wn * 32;

  const unsigned short* __restrict__ Ab = A + row0 * Kd;
  const unsigned short* __restrict__ Bb = B + (long)col0 * Kd;

  v8f acc00 = {}, acc01 = {}, acc10 = {}, acc11 = {};
  for (int k0 = 0; k0 < Kd; k0 += 32) {
    __builtin_prefetch(Ab + k0 + 64, 0, 0);  // next K panel -> global_prefetch
    __builtin_prefetch(Bb + k0 + 64, 0, 0);
    v16bf a0 = load_frag_a(Ab + k0, Kd, lane);
    v16bf a1 = load_frag_a(Ab + (long)16 * Kd + k0, Kd, lane);
    v16bf b0 = load_frag_b(Bb + k0, Kd, lane);
    v16bf b1 = load_frag_b(Bb + (long)16 * Kd + k0, Kd, lane);
    acc00 = wmma_bf16(a0, b0, acc00);
    acc01 = wmma_bf16(a0, b1, acc01);
    acc10 = wmma_bf16(a1, b0, acc10);
    acc11 = wmma_bf16(a1, b1, acc11);
  }

  const int hi = lane >> 4, nl = lane & 15;
  v8f* accs[4] = {&acc00, &acc01, &acc10, &acc11};
#pragma unroll
  for (int im = 0; im < 2; ++im) {
#pragma unroll
    for (int in = 0; in < 2; ++in) {
      v8f& c = *accs[im * 2 + in];
#pragma unroll
      for (int i = 0; i < 8; ++i) {
        const long m = row0 + im * 16 + hi * 8 + i;
        const int n = col0 + in * 16 + nl;
        float v = c[i] + bias[n];
        if (MODE == 1) {
          Of[m * Nn + n] = v + resid[m * Nn + n];
        } else if (MODE == 2) {
          v = 0.5f * v * (1.0f + erff(v * 0.70710678118654752f));
          Obf[m * Nn + n] = f2bf(v);
        } else {
          Obf[m * Nn + n] = f2bf(v);
        }
      }
    }
  }
}

// ---------------------------------------------------------------------------
// Windowed attention, one workgroup per (block p, head h); gather/scatter fused.
__global__ void __launch_bounds__(256) attn_kernel(const unsigned short* __restrict__ qkv,
                                                   const int* __restrict__ order,
                                                   unsigned short* __restrict__ ao) {
  __shared__ unsigned short qs[KW * HD];   // 8 KB
  __shared__ unsigned short ks[KW * HD];   // 8 KB
  __shared__ unsigned short vs[KW * HD];   // 8 KB
  __shared__ unsigned short Pl[KW * KW];   // 32 KB softmax probs (bf16)

  const int p = blockIdx.x, h = blockIdx.y;
  const int t = threadIdx.x;
  const int lane = t & 31, wave = t >> 5;
  const int hi = lane >> 4, nl = lane & 15;

#if USE_ASYNC_LDS
  // Gather q/k/v rows straight into LDS on the ASYNCcnt path:
  // 128 rows x 64B per matrix = 512 x b128 chunks; 2 chunks/thread/matrix.
  for (int i = t; i < KW * 4; i += 256) {
    const int r = i >> 2;
    const int c = (i & 3) * 8;  // 8 bf16 = 16 bytes
    const long grow = order[p * KW + r];
    const unsigned short* gb = qkv + grow * (3 * CH) + h * HD + c;
    __builtin_amdgcn_global_load_async_to_lds_b128(
        (v4i*)(gb), (v4i*)(&qs[r * HD + c]), 0, 0);
    __builtin_amdgcn_global_load_async_to_lds_b128(
        (v4i*)(gb + CH), (v4i*)(&ks[r * HD + c]), 0, 0);
    __builtin_amdgcn_global_load_async_to_lds_b128(
        (v4i*)(gb + 2 * CH), (v4i*)(&vs[r * HD + c]), 0, 0);
  }
  __builtin_amdgcn_s_wait_asynccnt(0);
  __syncthreads();
#else
  for (int i = t; i < KW * HD; i += 256) {
    const int r = i >> 5, dd = i & 31;
    const long grow = order[p * KW + r];
    const unsigned short* base = qkv + grow * (3 * CH) + h * HD + dd;
    qs[i] = base[0];
    ks[i] = base[CH];
    vs[i] = base[2 * CH];
  }
  __syncthreads();
#endif

  // scores: wave w owns rows [16w, 16w+16); one WMMA per 16x16 tile (d == K == 32)
  v8f sc[8];
  {
    const v16bf aq = load_frag_a(qs + wave * 16 * HD, HD, lane);
#pragma unroll
    for (int tj = 0; tj < 8; ++tj) {
      v16bf bk = load_frag_b(ks + tj * 16 * HD, HD, lane);
      v8f c = {};
      sc[tj] = wmma_bf16(aq, bk, c);
    }
  }

  // register softmax over this wave's 128 columns per row
#pragma unroll
  for (int i = 0; i < 8; ++i) {
    float mx = -3.0e38f;
#pragma unroll
    for (int tj = 0; tj < 8; ++tj) mx = fmaxf(mx, sc[tj][i] * SCALE);
#pragma unroll
    for (int m = 1; m < 16; m <<= 1) mx = fmaxf(mx, __shfl_xor(mx, m, 32));
    float e[8], sum = 0.0f;
#pragma unroll
    for (int tj = 0; tj < 8; ++tj) {
      e[tj] = __expf(sc[tj][i] * SCALE - mx);
      sum += e[tj];
    }
#pragma unroll
    for (int m = 1; m < 16; m <<= 1) sum += __shfl_xor(sum, m, 32);
    const float inv = 1.0f / sum;
    const int r = wave * 16 + hi * 8 + i;
#pragma unroll
    for (int tj = 0; tj < 8; ++tj) Pl[r * KW + tj * 16 + nl] = f2bf(e[tj] * inv);
  }
  __syncthreads();

  // O = P @ V : wave w computes its 16x32 output rows; K = 128 in 4 WMMA steps
  v8f o0 = {}, o1 = {};
#pragma unroll
  for (int k0 = 0; k0 < KW; k0 += 32) {
    v16bf ap = load_frag_a(Pl + wave * 16 * KW + k0, KW, lane);
    v16bf bv0 = load_frag_bT(vs + k0 * HD + 0, HD, lane);
    v16bf bv1 = load_frag_bT(vs + k0 * HD + 16, HD, lane);
    o0 = wmma_bf16(ap, bv0, o0);
    o1 = wmma_bf16(ap, bv1, o1);
  }

  // scatter (== out[inverse] in the reference)
#pragma unroll
  for (int i = 0; i < 8; ++i) {
    const int r = wave * 16 + hi * 8 + i;
    const long grow = order[p * KW + r];
    ao[grow * CH + h * HD + nl] = f2bf(o0[i]);
    ao[grow * CH + h * HD + 16 + nl] = f2bf(o1[i]);
  }
}

// ---------------------------------------------------------------------------
extern "C" void kernel_launch(void* const* d_in, const int* in_sizes, int n_in,
                              void* d_out, int out_size, void* d_ws, size_t ws_size,
                              hipStream_t stream) {
  (void)in_sizes; (void)n_in; (void)out_size; (void)ws_size;
  const float* feat   = (const float*)d_in[0];
  const int*   order  = (const int*)d_in[1];
  /* inverse (d_in[2]) not needed: scatter by `order` is equivalent */
  const float* qkv_w  = (const float*)d_in[3];
  const float* qkv_b  = (const float*)d_in[4];
  const float* proj_w = (const float*)d_in[5];
  const float* proj_b = (const float*)d_in[6];
  const float* ln1_g  = (const float*)d_in[7];
  const float* ln1_b  = (const float*)d_in[8];
  const float* ln2_g  = (const float*)d_in[9];
  const float* ln2_b  = (const float*)d_in[10];
  const float* mlp_w1 = (const float*)d_in[11];
  const float* mlp_b1 = (const float*)d_in[12];
  const float* mlp_w2 = (const float*)d_in[13];
  const float* mlp_b2 = (const float*)d_in[14];
  float* out = (float*)d_out;

  char* ws = (char*)d_ws;
  size_t off = 0;
  auto alloc = [&](size_t bytes) { char* p = ws + off; off += (bytes + 255) & ~size_t(255); return p; };
  unsigned short* qkv_w_bf = (unsigned short*)alloc((size_t)3 * CH * CH * 2);
  unsigned short* proj_w_bf = (unsigned short*)alloc((size_t)CH * CH * 2);
  unsigned short* w1_bf = (unsigned short*)alloc((size_t)HID * CH * 2);
  unsigned short* w2_bf = (unsigned short*)alloc((size_t)CH * HID * 2);
  unsigned short* x_bf = (unsigned short*)alloc((size_t)NTOK * CH * 2);   // LN1 out; reused for LN2 out
  unsigned short* ao_bf = (unsigned short*)alloc((size_t)NTOK * CH * 2);  // attention out
  unsigned short* big = (unsigned short*)alloc((size_t)NTOK * HID * 2);   // qkv (N x 768) then h (N x 1024)
  unsigned short* qkv_bf = big;
  unsigned short* h_bf = big;

  // 1) weights -> bf16
  cvt_bf16_kernel<<<(3 * CH * CH + 255) / 256, 256, 0, stream>>>(qkv_w, qkv_w_bf, 3 * CH * CH);
  cvt_bf16_kernel<<<(CH * CH + 255) / 256, 256, 0, stream>>>(proj_w, proj_w_bf, CH * CH);
  cvt_bf16_kernel<<<(HID * CH + 255) / 256, 256, 0, stream>>>(mlp_w1, w1_bf, HID * CH);
  cvt_bf16_kernel<<<(CH * HID + 255) / 256, 256, 0, stream>>>(mlp_w2, w2_bf, CH * HID);

  // 2) x = LN1(feat) -> bf16
  ln_bf16_kernel<<<NTOK, 256, 0, stream>>>(feat, ln1_g, ln1_b, x_bf);

  // 3) qkv = x @ qkv_w^T + qkv_b
  gemm_bf16_kernel<0><<<dim3((3 * CH) / 64, NTOK / 128), 256, 0, stream>>>(
      x_bf, qkv_w_bf, qkv_b, nullptr, qkv_bf, nullptr, 3 * CH, CH);

  // 4) windowed attention with fused gather/scatter
  attn_kernel<<<dim3(NTOK / KW, NH), 256, 0, stream>>>(qkv_bf, order, ao_bf);

  // 5) x = feat + ao @ proj_w^T + proj_b  -> f32 in d_out
  gemm_bf16_kernel<1><<<dim3(CH / 64, NTOK / 128), 256, 0, stream>>>(
      ao_bf, proj_w_bf, proj_b, feat, nullptr, out, CH, CH);

  // 6) y = LN2(x) -> bf16 (reuse x_bf)
  ln_bf16_kernel<<<NTOK, 256, 0, stream>>>(out, ln2_g, ln2_b, x_bf);

  // 7) h = gelu(y @ w1^T + b1) -> bf16 (reuses qkv region)
  gemm_bf16_kernel<2><<<dim3(HID / 64, NTOK / 128), 256, 0, stream>>>(
      x_bf, w1_bf, mlp_b1, nullptr, h_bf, nullptr, HID, CH);

  // 8) out = x + h @ w2^T + b2
  gemm_bf16_kernel<1><<<dim3(CH / 64, NTOK / 128), 256, 0, stream>>>(
      h_bf, w2_bf, mlp_b2, out, nullptr, out, CH, HID);
}
